// ChartParser_75840532512785
// MI455X (gfx1250) — matrix operand order repaired and checked by reading, not compiled, not run on hardware
//
#include <hip/hip_runtime.h>
#include <hip/hip_bf16.h>
#include <stddef.h>
#include <stdint.h>

// ---------------------------------------------------------------------------
// Types for CDNA5 WMMA (wave32): v_wmma_f32_16x16x32_bf16
// ---------------------------------------------------------------------------
typedef __attribute__((ext_vector_type(16))) __bf16 v16bf;
typedef __attribute__((ext_vector_type(8)))  float  v8f;
typedef __attribute__((ext_vector_type(4)))  unsigned int u32x4;

union FragBF { u32x4 q[2]; v16bf v; };

__device__ __forceinline__ unsigned short f32_to_bf16(float f) {
  unsigned int u = __builtin_bit_cast(unsigned int, f);
  u += 0x7fffu + ((u >> 16) & 1u);   // round-to-nearest-even
  return (unsigned short)(u >> 16);
}
__device__ __forceinline__ float bf16_to_f32(unsigned short s) {
  unsigned int u = ((unsigned int)s) << 16;
  return __builtin_bit_cast(float, u);
}

// ---------------------------------------------------------------------------
// Tiled WMMA GEMM:  C[M,N] = leaky( A[M,K] * B + colBias[n] + rowBias[m] )
//   A_F32 : A is fp32, else raw bf16 (ushort)
//   B_F32 : B is fp32, else raw bf16
//   B_NT  : B stored N x K row-major (i.e. compute A * B^T); else K x N
//   OUT_F32: C fp32, else bf16 (ushort)
//   ASYNC : double-buffered GLOBAL_LOAD_ASYNC_TO_LDS staging
//           (requires raw-bf16 A and B, B_NT layout)
// Block = 256 threads (8 waves). Block tile 128x128, K-step 32.
// Wave grid 4x2: each wave owns 32x64 => 2x4 WMMA accumulators.
// All problem dims must be multiples of the tile sizes (true here).
// ---------------------------------------------------------------------------
template<bool A_F32, bool B_F32, bool B_NT, bool OUT_F32, bool ASYNC>
__global__ __launch_bounds__(256) void gemm_wmma_kernel(
    const void* __restrict__ Av, int lda, long long aBatch,
    const void* __restrict__ Bv, int ldb, long long bBatch,
    void* __restrict__ Cv, int ldc, long long cBatch,
    const float* __restrict__ colBias,
    const float* __restrict__ rowBias, long long rbBatch,
    float slope, int K)
{
  constexpr int NBUF = ASYNC ? 2 : 1;
  __shared__ __align__(16) unsigned short sA[NBUF][128][40];   // [m][k], pad 8
  __shared__ __align__(16) unsigned short sB[NBUF][128][40];   // [n][k], pad 8

  const int z = blockIdx.z;
  const float*          Af = (const float*)Av + (size_t)z * (size_t)aBatch;
  const unsigned short* Ab = (const unsigned short*)Av + (size_t)z * (size_t)aBatch;
  const float*          Bf = (const float*)Bv + (size_t)z * (size_t)bBatch;
  const unsigned short* Bb = (const unsigned short*)Bv + (size_t)z * (size_t)bBatch;
  float*          Cf = (float*)Cv + (size_t)z * (size_t)cBatch;
  unsigned short* Cb = (unsigned short*)Cv + (size_t)z * (size_t)cBatch;
  const float* rB = rowBias ? (rowBias + (size_t)z * (size_t)rbBatch) : nullptr;

  const int tid  = threadIdx.x;
  const int bm   = blockIdx.y * 128;
  const int bn   = blockIdx.x * 128;
  const int lane = tid & 31;
  const int wave = tid >> 5;
  const int wm   = (wave & 3) * 32;   // wave row offset within block tile
  const int wn   = (wave >> 2) * 64;  // wave col offset within block tile
  const int half = lane >> 4;         // 0: lanes 0-15, 1: lanes 16-31
  const int lr   = lane & 15;

  v8f acc[2][4];
#pragma unroll
  for (int i = 0; i < 2; ++i)
#pragma unroll
    for (int j = 0; j < 4; ++j)
#pragma unroll
      for (int e = 0; e < 8; ++e) acc[i][j][e] = 0.0f;

  // ---- fragment load (ISA 16-bit A layout; same K layout for B) + WMMA ----
  auto compute_step = [&](const unsigned short (*pA)[40],
                          const unsigned short (*pB)[40]) {
    FragBF afr[2], bfr[4];
#pragma unroll
    for (int tm = 0; tm < 2; ++tm) {
      const int m = wm + tm * 16 + lr;
      afr[tm].q[0] = *(const u32x4*)&pA[m][half * 8];        // K = h*8 .. h*8+7
      afr[tm].q[1] = *(const u32x4*)&pA[m][16 + half * 8];   // K = 16+h*8 ..
    }
#pragma unroll
    for (int tn = 0; tn < 4; ++tn) {
      const int n = wn + tn * 16 + lr;
      bfr[tn].q[0] = *(const u32x4*)&pB[n][half * 8];        // K = h*8 .. h*8+7
      bfr[tn].q[1] = *(const u32x4*)&pB[n][16 + half * 8];   // K = 16+h*8 ..
    }
#pragma unroll
    for (int tm = 0; tm < 2; ++tm)
#pragma unroll
      for (int tn = 0; tn < 4; ++tn)
        acc[tm][tn] = __builtin_amdgcn_wmma_f32_16x16x32_bf16(
            false, afr[tm].v, false, bfr[tn].v, (short)0, acc[tm][tn],
            false, false);
  };

  if constexpr (ASYNC) {
    // ---- double-buffered async global->LDS staging (raw bf16 tiles) ----
    // Per instruction: every lane copies 16 B; 256 threads cover 64 rows
    // (row = tid>>2, 16-B chunk = (tid&3)); two issues cover the 128-row tile.
    const int ar = tid >> 2;          // 0..63
    const int ac = (tid & 3) * 8;     // ushort offset: 0,8,16,24 (16 B chunks)
    auto issue_tile = [&](const unsigned short* g, int ld, int base, int k0,
                          unsigned short (*dst)[40]) {
#pragma unroll
      for (int p = 0; p < 2; ++p) {
        const unsigned short* gp =
            g + (size_t)(base + ar + 64 * p) * (size_t)ld + (size_t)(k0 + ac);
        const unsigned ldsOff =
            (unsigned)(uintptr_t)(const void*)&dst[ar + 64 * p][ac];
        asm volatile("global_load_async_to_lds_b128 %0, %1, off"
                     :: "v"(ldsOff),
                        "v"((unsigned long long)(uintptr_t)gp)
                     : "memory");
      }
    };

    issue_tile(Ab, lda, bm, 0, sA[0]);   // prologue: fill buffer 0
    issue_tile(Bb, ldb, bn, 0, sB[0]);

    for (int k0 = 0; k0 < K; k0 += 32) {
      const int cur = (k0 >> 5) & 1;
      const bool more = (k0 + 32) < K;
      if (more) {
        // prefetch next tile into the other buffer (reads of it finished at
        // the trailing barrier of the previous iteration)
        issue_tile(Ab, lda, bm, k0 + 32, sA[cur ^ 1]);
        issue_tile(Bb, ldb, bn, k0 + 32, sB[cur ^ 1]);
        // 4 newer loads in flight; in-order completion => current buffer done
        asm volatile("s_wait_asynccnt 0x4" ::: "memory");
      } else {
        asm volatile("s_wait_asynccnt 0x0" ::: "memory");
      }
      __syncthreads();                  // make all waves' LDS writes visible
      compute_step(sA[cur], sB[cur]);
      __syncthreads();                  // frags in VGPRs; buffer reusable
    }
  } else {
    // ---- synchronous staging with fp32->bf16 conversion + prefetch ----
    for (int k0 = 0; k0 < K; k0 += 32) {
      const bool more = (k0 + 32) < K;
      // A tile (128 x 32)
#pragma unroll
      for (int p = 0; p < 4; ++p) {
        const int r = (tid >> 3) + p * 32;
        const int c = (tid & 7) * 4;
        const size_t g = (size_t)(bm + r) * (size_t)lda + (size_t)(k0 + c);
        if constexpr (A_F32) {
          if (more) __builtin_prefetch(&Af[g + 32], 0, 1);
#pragma unroll
          for (int u = 0; u < 4; ++u) sA[0][r][c + u] = f32_to_bf16(Af[g + u]);
        } else {
          if (more) __builtin_prefetch(&Ab[g + 32], 0, 1);
#pragma unroll
          for (int u = 0; u < 4; ++u) sA[0][r][c + u] = Ab[g + u];
        }
      }
      // B tile -> LDS transposed: sB[n][k]
      if constexpr (B_NT) {
#pragma unroll
        for (int p = 0; p < 4; ++p) {
          const int r = (tid >> 3) + p * 32;   // n
          const int c = (tid & 7) * 4;         // k
          const size_t g = (size_t)(bn + r) * (size_t)ldb + (size_t)(k0 + c);
#pragma unroll
          for (int u = 0; u < 4; ++u) {
            if constexpr (B_F32) sB[0][r][c + u] = f32_to_bf16(Bf[g + u]);
            else                 sB[0][r][c + u] = Bb[g + u];
          }
        }
      } else {
        const int kk = tid >> 5;          // 0..7
        const int n4 = (tid & 31) * 4;    // 0..124
#pragma unroll
        for (int p = 0; p < 4; ++p) {
          const int c = kk + p * 8;       // k within tile
          const size_t g = (size_t)(k0 + c) * (size_t)ldb + (size_t)(bn + n4);
          if (more) __builtin_prefetch(&Bf[g + 32 * (size_t)ldb], 0, 1);
#pragma unroll
          for (int u = 0; u < 4; ++u) {
            if constexpr (B_F32) sB[0][n4 + u][c] = f32_to_bf16(Bf[g + u]);
            else                 sB[0][n4 + u][c] = Bb[g + u];
          }
        }
      }
      __syncthreads();
      compute_step(sA[0], sB[0]);
      __syncthreads();
    }
  }

  // ---- epilogue: bias + leaky + store ----
#pragma unroll
  for (int tm = 0; tm < 2; ++tm)
#pragma unroll
    for (int tn = 0; tn < 4; ++tn) {
      const int gn = bn + wn + tn * 16 + lr;
      const float cb = colBias ? colBias[gn] : 0.0f;
#pragma unroll
      for (int v = 0; v < 8; ++v) {
        const int gm = bm + wm + tm * 16 + half * 8 + v;
        float val = acc[tm][tn][v] + cb;
        if (rB) val += rB[gm];
        val = (val >= 0.0f) ? val : val * slope;
        if constexpr (OUT_F32) Cf[(size_t)gm * (size_t)ldc + gn] = val;
        else                   Cb[(size_t)gm * (size_t)ldc + gn] = f32_to_bf16(val);
      }
    }
}

// ---------------------------------------------------------------------------
// lc[r] = lefts[r,:] . Wbil[:1024,1024] + Wbil[1024,1024] + bbil
// ---------------------------------------------------------------------------
__global__ void lc_kernel(const unsigned short* __restrict__ leftsb,
                          const float* __restrict__ Wbil,
                          const float* __restrict__ bbil,
                          float* __restrict__ lc)
{
  const int r = blockIdx.x * blockDim.x + threadIdx.x;
  if (r >= 64 * 256) return;
  const unsigned short* row = leftsb + (size_t)r * 1024;
  float acc = 0.0f;
  for (int h = 0; h < 1024; ++h)
    acc += bf16_to_f32(row[h]) * Wbil[(size_t)h * 1025 + 1024];
  lc[r] = acc + Wbil[(size_t)1024 * 1025 + 1024] + bbil[0];
}

// ---------------------------------------------------------------------------
// CKY over one 256x256 score matrix per block; chart lives entirely in LDS
// (256 KB of the WGP's 320 KB). split stored as u8 in global.
// ---------------------------------------------------------------------------
__global__ __launch_bounds__(256) void cky_kernel(
    const float* __restrict__ scores, unsigned char* __restrict__ split)
{
  extern __shared__ float chart[];          // 256*256 floats = 256 KB
  const int b = blockIdx.x;
  const int i = threadIdx.x;                // start position
  const float* S = scores + (size_t)b * 65536;
  unsigned char* sp = split + (size_t)b * 65536;

  for (int idx = i; idx < 65536; idx += 256) chart[idx] = 0.0f;
  __syncthreads();

  for (int w = 1; w < 256; ++w) {
    const int j = i + w;
    if (j < 256) {
      const float* rowi = chart + i * 256;
      float bestv = -3.0e38f;
      int bestm = 0;
      for (int m = 0; m < w; ++m) {
        const float v = rowi[i + m] + chart[(i + m + 1) * 256 + j];
        if (v > bestv) { bestv = v; bestm = m; }   // first-max, like argmax
      }
      chart[i * 256 + j] = S[i * 256 + j] + bestv;
      sp[i * 256 + j] = (unsigned char)(i + bestm);
    }
    __syncthreads();                         // width-w entries visible for w+1
  }
}

// ---------------------------------------------------------------------------
// Backtrack best tree per batch; writes zero-padded (lefts, rights) as float.
// d_out layout: [B*S lefts][B*S rights].
// ---------------------------------------------------------------------------
__global__ void backtrack_kernel(const unsigned char* __restrict__ split,
                                 float* __restrict__ out)
{
  const int b = blockIdx.x;
  const unsigned char* sp = split + (size_t)b * 65536;
  float* lefts  = out + (size_t)b * 256;
  float* rights = out + (size_t)64 * 256 + (size_t)b * 256;
  for (int k = threadIdx.x; k < 256; k += blockDim.x) {
    lefts[k] = 0.0f;
    rights[k] = 0.0f;
  }
  __syncthreads();
  if (threadIdx.x == 0) {
    short stI[512], stJ[512];
    int top = 0, cnt = 0;
    stI[0] = 0; stJ[0] = 255; top = 1;
    for (int t = 0; t < 255; ++t) {
      if (top > 0) {
        const int i = stI[top - 1], j = stJ[top - 1];
        --top;
        lefts[cnt] = (float)i;
        rights[cnt] = (float)j;
        ++cnt;
        const int s = sp[i * 256 + j];
        stI[top] = (short)i;       stJ[top] = (short)s; if (s > i)     ++top;
        stI[top] = (short)(s + 1); stJ[top] = (short)j; if (j > s + 1) ++top;
      }
    }
  }
}

// ---------------------------------------------------------------------------
// Orchestration
// ---------------------------------------------------------------------------
extern "C" void kernel_launch(void* const* d_in, const int* in_sizes, int n_in,
                              void* d_out, int out_size, void* d_ws, size_t ws_size,
                              hipStream_t stream) {
  (void)in_sizes; (void)n_in; (void)out_size; (void)ws_size;
  const float* X    = (const float*)d_in[0];   // [64,256,1024]
  // d_in[1] = mask (unused by reference)
  const float* Wl   = (const float*)d_in[2];   // [1024,1024]
  const float* bl   = (const float*)d_in[3];   // [1024]
  const float* Wr   = (const float*)d_in[4];
  const float* br   = (const float*)d_in[5];
  const float* Wbil = (const float*)d_in[6];   // [1025,1025]
  const float* bbil = (const float*)d_in[7];   // scalar
  float* out = (float*)d_out;

  // workspace layout (~116 MB)
  char* ws = (char*)d_ws;
  unsigned short* leftsb  = (unsigned short*)ws;                       // 16384x1024 bf16
  unsigned short* rightsb = leftsb  + (size_t)16384 * 1024;            // 16384x1024 bf16
  unsigned short* lwb     = rightsb + (size_t)16384 * 1024;            // 16384x1024 bf16
  float*          lc      = (float*)(lwb + (size_t)16384 * 1024);      // 16384 f32
  float*          scores  = lc + 16384;                                // 64x256x256 f32
  unsigned char*  split   = (unsigned char*)(scores + (size_t)64 * 256 * 256);

  const dim3 blk(256);

  // lefts = leaky(X @ Wl + bl), rights = leaky(X @ Wr + br)
  gemm_wmma_kernel<true, true, false, false, false>
      <<<dim3(8, 128, 1), blk, 0, stream>>>(
      X, 1024, 0, Wl, 1024, 0, leftsb, 1024, 0, bl, nullptr, 0, 0.1f, 1024);
  gemm_wmma_kernel<true, true, false, false, false>
      <<<dim3(8, 128, 1), blk, 0, stream>>>(
      X, 1024, 0, Wr, 1024, 0, rightsb, 1024, 0, br, nullptr, 0, 0.1f, 1024);

  // LWh = lefts @ Wbil[:1024,:1024] + Wbil[1024,:1024]   (ldb = 1025!)
  gemm_wmma_kernel<false, true, false, false, false>
      <<<dim3(8, 128, 1), blk, 0, stream>>>(
      leftsb, 1024, 0, Wbil, 1025, 0, lwb, 1024, 0,
      Wbil + (size_t)1024 * 1025, nullptr, 0, 1.0f, 1024);

  // lc = lefts @ Wbil[:1024,1024] + Wbil[1024,1024] + bbil
  lc_kernel<<<64, 256, 0, stream>>>(leftsb, Wbil, bbil, lc);

  // scores[b] = LWh[b] @ rights[b]^T + lc[b]
  // (per-batch, raw bf16 operands -> async double-buffered staging)
  gemm_wmma_kernel<false, false, true, true, true>
      <<<dim3(2, 2, 64), blk, 0, stream>>>(
      lwb, 1024, 256 * 1024, rightsb, 1024, 256 * 1024,
      scores, 256, 256 * 256, nullptr, lc, 256, 1.0f, 1024);

  // CKY DP: chart (256 KB) entirely in LDS per workgroup
  cky_kernel<<<64, 256, 256 * 256 * 4, stream>>>(scores, split);

  // Backtrack + zero-pad outputs
  backtrack_kernel<<<64, 64, 0, stream>>>(split, out);
}
